// RandomShift_59219009077860
// MI455X (gfx1250) — matrix operand order repaired and checked by reading, not compiled, not run on hardware
//
#include <hip/hip_runtime.h>
#include <stdint.h>

// Problem constants (from reference)
#define BB 65536
#define EE 512
#define NOISE_SCALE 0.01f

typedef __attribute__((ext_vector_type(16))) _Float16 v16h;
typedef __attribute__((ext_vector_type(8)))  _Float16 v8h;
typedef __attribute__((ext_vector_type(8)))  float    v8f;

#if __has_builtin(__builtin_amdgcn_tensor_load_to_lds)
#define HAVE_TDM 1
typedef __attribute__((ext_vector_type(4))) unsigned int u32x4;
typedef __attribute__((ext_vector_type(8))) int          i32x8;
typedef __attribute__((ext_vector_type(4))) int          i32x4;
#endif

// LDS staging layout: 32 rows of H, each 512 f16 = 1024B, padded by 16B
// -> row stride 1040B = 260 dwords == 4 (mod 64 banks): conflict-free b128 reads.
#define LDS_ROW_BYTES 1040
#define LDS_BYTES (32 * LDS_ROW_BYTES)

// ---------------------------------------------------------------------------
// Threefry-2x32 (20 rounds), matches JAX's threefry_2x32.
// ---------------------------------------------------------------------------
__device__ __forceinline__ void threefry2x32(uint32_t k0, uint32_t k1,
                                             uint32_t x0, uint32_t x1,
                                             uint32_t& o0, uint32_t& o1) {
  const uint32_t ks2 = 0x1BD11BDAu ^ k0 ^ k1;
  uint32_t X0 = x0 + k0;
  uint32_t X1 = x1 + k1;
#define TF_ROUND(R) { X0 += X1; X1 = (X1 << (R)) | (X1 >> (32 - (R))); X1 ^= X0; }
  TF_ROUND(13) TF_ROUND(15) TF_ROUND(26) TF_ROUND(6)
  X0 += k1;  X1 += ks2 + 1u;
  TF_ROUND(17) TF_ROUND(29) TF_ROUND(16) TF_ROUND(24)
  X0 += ks2; X1 += k0 + 2u;
  TF_ROUND(13) TF_ROUND(15) TF_ROUND(26) TF_ROUND(6)
  X0 += k0;  X1 += k1 + 3u;
  TF_ROUND(17) TF_ROUND(29) TF_ROUND(16) TF_ROUND(24)
  X0 += k1;  X1 += ks2 + 4u;
  TF_ROUND(13) TF_ROUND(15) TF_ROUND(26) TF_ROUND(6)
  X0 += ks2; X1 += k0 + 5u;
#undef TF_ROUND
  o0 = X0; o1 = X1;
}

__device__ __forceinline__ float erfinv_f32(float x) {
  float w = -__logf((1.0f - x) * (1.0f + x));
  float p;
  if (w < 5.0f) {
    w = w - 2.5f;
    p = 2.81022636e-08f;
    p = fmaf(p, w, 3.43273939e-07f);
    p = fmaf(p, w, -3.5233877e-06f);
    p = fmaf(p, w, -4.39150654e-06f);
    p = fmaf(p, w, 0.00021858087f);
    p = fmaf(p, w, -0.00125372503f);
    p = fmaf(p, w, -0.00417768164f);
    p = fmaf(p, w, 0.246640727f);
    p = fmaf(p, w, 1.50140941f);
  } else {
    w = __fsqrt_rn(w) - 3.0f;
    p = -0.000200214257f;
    p = fmaf(p, w, 0.000100950558f);
    p = fmaf(p, w, 0.00134934322f);
    p = fmaf(p, w, -0.00367342844f);
    p = fmaf(p, w, 0.00573950773f);
    p = fmaf(p, w, -0.0076224613f);
    p = fmaf(p, w, 0.00943887047f);
    p = fmaf(p, w, 1.00167406f);
    p = fmaf(p, w, 2.83297682f);
  }
  return p * x;
}

__device__ __forceinline__ float bits_to_normal(uint32_t bits) {
  float f = (float)(bits >> 9) * (1.0f / 8388608.0f);   // [0,1)
  float u = fmaf(f, 1.99999994f, -0.99999994f);         // (-1, 1)
  return 1.41421356237f * erfinv_f32(u);
}

// ---------------------------------------------------------------------------
// Kernel 1: convert W (f32 [E,E]) -> f16
// ---------------------------------------------------------------------------
__global__ __launch_bounds__(256) void convert_w_kernel(const float* __restrict__ W,
                                                        _Float16* __restrict__ W16) {
  int idx = blockIdx.x * 256 + threadIdx.x;
  W16[idx] = (_Float16)W[idx];
}

// ---------------------------------------------------------------------------
// Kernel 2: H[i,:] = f16( x[i,:] + 0.01 * normals(fold_in(key0, label[i])) )
// ---------------------------------------------------------------------------
__global__ __launch_bounds__(256) void noise_shift_kernel(const float* __restrict__ x,
                                                          const long long* __restrict__ label,
                                                          _Float16* __restrict__ H) {
  const int row = blockIdx.x;
  const int j   = threadIdx.x;
  const long long lab = label[row];
  const uint32_t lab_hi = (uint32_t)((uint64_t)lab >> 32);
  const uint32_t lab_lo = (uint32_t)(uint64_t)lab;

  uint32_t nk0, nk1;
  threefry2x32(0u, 0u, lab_hi, lab_lo, nk0, nk1);

  uint32_t b0, b1;
  threefry2x32(nk0, nk1, (uint32_t)j, (uint32_t)(j + 256), b0, b1);

  const float n0 = bits_to_normal(b0);
  const float n1 = bits_to_normal(b1);

  const float* xr = x + (size_t)row * EE;
  _Float16*    hr = H + (size_t)row * EE;
  hr[j]       = (_Float16)fmaf(NOISE_SCALE, n0, xr[j]);
  hr[j + 256] = (_Float16)fmaf(NOISE_SCALE, n1, xr[j + 256]);
}

// ---------------------------------------------------------------------------
// Kernel 3: out = H @ W16^T + b.
// Block: 256 thr = 8 waves; owns 32 rows. Wave w owns cols [64w, 64w+64):
// 2 A-frags (from LDS) x 4 B-frags (global, double-buffered) -> 8 WMMA / K-step.
// A strip staged to LDS once per block by the TDM (tensor_load_to_lds) with
// 16B/row padding; fallback: cooperative copy.
// ---------------------------------------------------------------------------
__global__ __launch_bounds__(256) void gemm_wmma_kernel(const _Float16* __restrict__ H,
                                                        const _Float16* __restrict__ W16,
                                                        const float* __restrict__ bias,
                                                        float* __restrict__ out) {
  __shared__ alignas(128) char lds_a[LDS_BYTES];

  const int lane  = threadIdx.x & 31;
  const int wave  = threadIdx.x >> 5;      // 0..7
  const int i0    = blockIdx.x << 5;       // 32-row strip
  const int n0    = wave << 6;             // 64-col strip
  const int mrow  = lane & 15;
  const int khalf = lane >> 4;             // 0 or 1

  // ---- Stage A strip (32 x 512 f16, padded rows) into LDS ----
#ifdef HAVE_TDM
  if (wave == 0) {
    const uint64_t gaddr  = (uint64_t)(uintptr_t)(H + (size_t)i0 * EE); // tile start (bytes)
    const uint32_t ldsoff = (uint32_t)(uintptr_t)lds_a;                 // LDS byte offset
    u32x4 g0;
    g0[0] = 1u;                                                  // count=1 (valid user D#)
    g0[1] = ldsoff;                                              // lds_addr
    g0[2] = (uint32_t)gaddr;                                     // global_addr[31:0]
    g0[3] = (uint32_t)((gaddr >> 32) & 0x01FFFFFFu) | 0x80000000u; // addr[56:32] | type=2
    i32x8 g1;
    g1[0] = (int)((1u << 16)      // data_size = 2B
                | (1u << 20)      // pad_enable
                | (7u << 22)      // pad_interval: 256 dwords (=1024B row)
                | (3u << 25));    // pad_amount: 4 dwords (=16B)
    g1[1] = (int)(512u << 16);    // tensor_dim0 = 512
    g1[2] = (int)(32u << 16);     // tensor_dim1 = 32
    g1[3] = (int)(512u << 16);    // tile_dim0 = 512
    g1[4] = (int)32u;             // tile_dim1 = 32, tile_dim2 = 0
    g1[5] = (int)512;             // tensor_dim0_stride = 512
    g1[6] = 0;
    g1[7] = 0;
    i32x4 g2 = {0, 0, 0, 0};
    i32x4 g3 = {0, 0, 0, 0};
#if __clang_major__ >= 23
    i32x8 g4 = {0, 0, 0, 0, 0, 0, 0, 0};   // trailing descriptor words (unused; zero)
    __builtin_amdgcn_tensor_load_to_lds(g0, g1, g2, g3, g4, 0);
#else
    __builtin_amdgcn_tensor_load_to_lds(g0, g1, g2, g3, 0);
#endif
    __builtin_amdgcn_s_wait_tensorcnt(0);
  }
#else
  {
    const _Float16* hblk = H + (size_t)i0 * EE;
    for (int idx = threadIdx.x; idx < 32 * EE; idx += 256) {
      const int r = idx >> 9;
      const int c = idx & (EE - 1);
      *(_Float16*)(lds_a + r * LDS_ROW_BYTES + 2 * c) = hblk[r * EE + c];
    }
  }
#endif
  __syncthreads();

  const _Float16* wb = W16 + (size_t)(n0 + mrow) * EE + 8 * khalf;
  const int a_k_byte = 2 * (8 * khalf);   // base K byte offset for this lane

  auto loadA = [&](int mt, int k0) -> v16h {
    const char* p = lds_a + (16 * mt + mrow) * LDS_ROW_BYTES + 2 * k0 + a_k_byte;
    v8h lo = *(const v8h*)(p);
    v8h hi = *(const v8h*)(p + 32);       // K += 16
    return __builtin_shufflevector(lo, hi, 0, 1, 2, 3, 4, 5, 6, 7,
                                           8, 9, 10, 11, 12, 13, 14, 15);
  };
  auto loadB = [&](int nt, int k0) -> v16h {
    const _Float16* q = wb + (size_t)nt * 16 * EE + k0;
    v8h lo = *(const v8h*)(q);
    v8h hi = *(const v8h*)(q + 16);
    return __builtin_shufflevector(lo, hi, 0, 1, 2, 3, 4, 5, 6, 7,
                                           8, 9, 10, 11, 12, 13, 14, 15);
  };
#define WMMA(A, Bm, C) __builtin_amdgcn_wmma_f32_16x16x32_f16(false, (A), false, (Bm), (short)0, (C), false, false)

  v8f c00 = {}, c01 = {}, c02 = {}, c03 = {};
  v8f c10 = {}, c11 = {}, c12 = {}, c13 = {};

  // Double-buffered B fragments (global); A fragments from LDS.
  v16h b0 = loadB(0, 0), b1 = loadB(1, 0), b2 = loadB(2, 0), b3 = loadB(3, 0);

  for (int k0 = 0; k0 < EE; k0 += 32) {
    const int kn = (k0 + 32 < EE) ? (k0 + 32) : 0;  // wrap: redundant but branch-free
    v16h nb0 = loadB(0, kn);
    v16h nb1 = loadB(1, kn);
    v16h nb2 = loadB(2, kn);
    v16h nb3 = loadB(3, kn);

    v16h a0 = loadA(0, k0);
    v16h a1 = loadA(1, k0);

    c00 = WMMA(a0, b0, c00);
    c01 = WMMA(a0, b1, c01);
    c02 = WMMA(a0, b2, c02);
    c03 = WMMA(a0, b3, c03);
    c10 = WMMA(a1, b0, c10);
    c11 = WMMA(a1, b1, c11);
    c12 = WMMA(a1, b2, c12);
    c13 = WMMA(a1, b3, c13);

    b0 = nb0; b1 = nb1; b2 = nb2; b3 = nb3;
  }
#undef WMMA

  const float bv0 = bias[n0 + mrow];
  const float bv1 = bias[n0 + 16 + mrow];
  const float bv2 = bias[n0 + 32 + mrow];
  const float bv3 = bias[n0 + 48 + mrow];

  // C layout: lane L, vgpr r -> row = r + 8*(L/16), col = L%16.
#pragma unroll
  for (int mt = 0; mt < 2; ++mt) {
    float* ob = out + (size_t)(i0 + 16 * mt + 8 * khalf) * EE + n0 + mrow;
    const v8f& d0 = mt ? c10 : c00;
    const v8f& d1 = mt ? c11 : c01;
    const v8f& d2 = mt ? c12 : c02;
    const v8f& d3 = mt ? c13 : c03;
#pragma unroll
    for (int r = 0; r < 8; ++r) {
      ob[(size_t)r * EE + 0]  = d0[r] + bv0;
      ob[(size_t)r * EE + 16] = d1[r] + bv1;
      ob[(size_t)r * EE + 32] = d2[r] + bv2;
      ob[(size_t)r * EE + 48] = d3[r] + bv3;
    }
  }
}

// ---------------------------------------------------------------------------
// Host entry
// ---------------------------------------------------------------------------
extern "C" void kernel_launch(void* const* d_in, const int* in_sizes, int n_in,
                              void* d_out, int out_size, void* d_ws, size_t ws_size,
                              hipStream_t stream) {
  const float*     x     = (const float*)d_in[0];
  const long long* label = (const long long*)d_in[1];
  const float*     W     = (const float*)d_in[2];
  const float*     bias  = (const float*)d_in[3];
  float*           out   = (float*)d_out;

  _Float16* H   = (_Float16*)d_ws;
  _Float16* W16 = (_Float16*)((char*)d_ws + (size_t)BB * EE * sizeof(_Float16));

  convert_w_kernel<<<(EE * EE) / 256, 256, 0, stream>>>(W, W16);
  noise_shift_kernel<<<BB, 256, 0, stream>>>(x, label, H);
  gemm_wmma_kernel<<<BB / 32, 256, 0, stream>>>(H, W16, bias, out);
}
